// GNNLayer_10531259810483
// MI455X (gfx1250) — matrix operand order repaired and checked by reading, not compiled, not run on hardware
//
#include <hip/hip_runtime.h>
#include <hip/hip_bf16.h>

// ---------------------------------------------------------------------------
// GNN layer for MI455X (gfx1250): all GEMMs on v_wmma_f32_16x16x32_f16,
// BN folded into the following GEMM's weights, f16 edge activations,
// gather-into-WMMA-fragments for x[src]/x[dst], atomic scatter for segment_sum.
// ---------------------------------------------------------------------------

constexpr int Nn = 50000;
constexpr int Ee = 300000;
constexpr int Hh = 128;

typedef __attribute__((ext_vector_type(16))) _Float16 v16h;
typedef __attribute__((ext_vector_type(8)))  _Float16 v8h;
typedef __attribute__((ext_vector_type(8)))  float    v8f;

#define WMMA_F16(acc, a, b) \
  (acc) = __builtin_amdgcn_wmma_f32_16x16x32_f16(false, (a), false, (b), (short)0, (acc), false, false)

__device__ __forceinline__ v8f bcast8(float x) {
  v8f r;
#pragma unroll
  for (int i = 0; i < 8; ++i) r[i] = x;
  return r;
}

__device__ __forceinline__ v16h mk_frag(v8h lo, v8h hi) {
  v16h r;
#pragma unroll
  for (int i = 0; i < 8; ++i) { r[i] = lo[i]; r[i + 8] = hi[i]; }
  return r;
}

// A fragment (16x32 f16, M x K) from a row-major f16 buffer.
// rowbase already points at this lane's row (row = lane&15).
// Lanes 0-15: K = kk+0..7 and kk+16..23 ; lanes 16-31: K = kk+8..15 and kk+24..31.
__device__ __forceinline__ v16h a_frag(const _Float16* rowbase, int kk, int lane) {
  const _Float16* p = rowbase + kk + ((lane >> 4) << 3);
  return mk_frag(*(const v8h*)p, *(const v8h*)(p + 16));
}

// A fragment from a row-major f32 buffer (convert on the fly).
__device__ __forceinline__ v16h a_frag_f32(const float* rowbase, int kk, int lane) {
  const float* p = rowbase + kk + ((lane >> 4) << 3);
  v16h r;
#pragma unroll
  for (int i = 0; i < 8; ++i) { r[i] = (_Float16)p[i]; r[i + 8] = (_Float16)p[i + 16]; }
  return r;
}

// B fragment (32x16 f16, K x N) from weights stored n-major: wbt[n][k], stride ld.
// Lanes 0-15: N = n0+lane, K = kk+0..7 and kk+16..23 ; lanes 16-31: N = n0+lane-16, K = kk+8..15 and kk+24..31.
__device__ __forceinline__ v16h b_frag(const _Float16* wbt, int ld, int n0, int kk, int lane) {
  const _Float16* p = wbt + (size_t)(n0 + (lane & 15)) * ld + kk + ((lane >> 4) << 3);
  return mk_frag(*(const v8h*)p, *(const v8h*)(p + 16));
}

// ---------------------------------------------------------------------------
// prep: W (in_dim x 128, row-major f32) -> bt (128 x in_pad, f16, zero padded)
__global__ __launch_bounds__(128) void k_prep_w(const float* __restrict__ W,
                                                _Float16* __restrict__ bt,
                                                int in_dim, int in_pad) {
  int n = blockIdx.x;
  for (int k = threadIdx.x; k < in_pad; k += 128)
    bt[(size_t)n * in_pad + k] = (_Float16)(k < in_dim ? W[(size_t)k * Hh + n] : 0.0f);
}

__global__ void k_f32_to_f16(const float* __restrict__ in, _Float16* __restrict__ out, int n) {
  int i = blockIdx.x * blockDim.x + threadIdx.x;
  if (i < n) out[i] = (_Float16)in[i];
}

// ---------------------------------------------------------------------------
// Edge stage 1: G = RBF(edge_attr) (16x224 per wave, LDS), gt = G@Wb + bb,
// a1 = lrelu(x[src]@We1_top + gt@We1_mid + x[dst]@We1_bot + be1); stats over cols.
__global__ __launch_bounds__(128) void k_edge1(
    const _Float16* __restrict__ xh, const float* __restrict__ ea,
    const int* __restrict__ ei,
    const _Float16* __restrict__ Wbbt, const float* __restrict__ bb,
    const _Float16* __restrict__ We1bt, const float* __restrict__ be1,
    _Float16* __restrict__ a1, float* __restrict__ gstats) {
  __shared__ _Float16 Gs[4][16 * 224];
  __shared__ float eaS[4][16 * 4];
  __shared__ float statsS[256];
  const int tid = threadIdx.x;
  const int wv = tid >> 5, lane = tid & 31;
  for (int i = tid; i < 256; i += 128) statsS[i] = 0.0f;
  __syncthreads();
  const int tile = blockIdx.x * 4 + wv;
  if (tile < (Ee / 16)) {
    if (lane < 16) {
      int e = tile * 16 + lane;
#pragma unroll
      for (int f = 0; f < 4; ++f) eaS[wv][lane * 4 + f] = ea[(size_t)e * 4 + f];
    }
    asm volatile("s_wait_dscnt 0" ::: "memory");
    // Build Gaussian expansion, pad cols 200..223 with zeros.
    for (int idx = lane; idx < 16 * 224; idx += 32) {
      int r = idx / 224, c = idx % 224;
      float g = 0.0f;
      if (c < 200) {
        int f = c / 50, ki = c % 50;
        float d = eaS[wv][r * 4 + f] - (float)ki * (1.0f / 49.0f);
        g = __expf(-d * d * 1250.0f);  // 1/(2*(1/50)^2)
      }
      Gs[wv][r * 224 + c] = (_Float16)g;
    }
    asm volatile("s_wait_dscnt 0" ::: "memory");

    const int col0 = lane & 15;
    const _Float16* grow = &Gs[wv][col0 * 224];
    v8f acc[8];
#pragma unroll
    for (int nt = 0; nt < 8; ++nt) acc[nt] = bcast8(bb[nt * 16 + col0]);
    for (int kk = 0; kk < 7; ++kk) {
      v16h a = a_frag(grow, kk * 32, lane);
#pragma unroll
      for (int nt = 0; nt < 8; ++nt) WMMA_F16(acc[nt], a, b_frag(Wbbt, 224, nt * 16, kk * 32, lane));
    }
    // Write gt back into Gs cols 0..127 (f16), D layout -> row-major.
    {
      int rb = (lane >> 4) << 3;
#pragma unroll
      for (int nt = 0; nt < 8; ++nt)
#pragma unroll
        for (int i = 0; i < 8; ++i)
          Gs[wv][(rb + i) * 224 + nt * 16 + col0] = (_Float16)acc[nt][i];
    }
    asm volatile("s_wait_dscnt 0" ::: "memory");

    // Gather rows of xh for src/dst of this lane's edge.
    int erow = tile * 16 + col0;
    const _Float16* xsrow = xh + (size_t)ei[erow] * Hh;
    const _Float16* xdrow = xh + (size_t)ei[Ee + erow] * Hh;

#pragma unroll
    for (int nt = 0; nt < 8; ++nt) acc[nt] = bcast8(be1[nt * 16 + col0]);
    for (int kk = 0; kk < 4; ++kk) {            // x[src] : We1 rows 0..127
      v16h a = a_frag(xsrow, kk * 32, lane);
#pragma unroll
      for (int nt = 0; nt < 8; ++nt) WMMA_F16(acc[nt], a, b_frag(We1bt, 384, nt * 16, kk * 32, lane));
    }
    for (int kk = 0; kk < 4; ++kk) {            // gt : We1 rows 128..255
      v16h a = a_frag(grow, kk * 32, lane);
#pragma unroll
      for (int nt = 0; nt < 8; ++nt) WMMA_F16(acc[nt], a, b_frag(We1bt, 384, nt * 16, 128 + kk * 32, lane));
    }
    for (int kk = 0; kk < 4; ++kk) {            // x[dst] : We1 rows 256..383
      v16h a = a_frag(xdrow, kk * 32, lane);
#pragma unroll
      for (int nt = 0; nt < 8; ++nt) WMMA_F16(acc[nt], a, b_frag(We1bt, 384, nt * 16, 256 + kk * 32, lane));
    }
    int rb = tile * 16 + ((lane >> 4) << 3);
#pragma unroll
    for (int nt = 0; nt < 8; ++nt) {
      int col = nt * 16 + col0;
      float ps = 0.0f, pss = 0.0f;
#pragma unroll
      for (int i = 0; i < 8; ++i) {
        float z = acc[nt][i];
        z = z >= 0.0f ? z : 0.01f * z;
        ps += z; pss += z * z;
        a1[(size_t)(rb + i) * Hh + col] = (_Float16)z;
      }
      atomicAdd(&statsS[col], ps);
      atomicAdd(&statsS[128 + col], pss);
    }
  }
  __syncthreads();
  for (int i = tid; i < 256; i += 128) atomicAdd(&gstats[i], statsS[i]);
}

// ---------------------------------------------------------------------------
// Generic: out = lrelu(in @ wbt + bias); per-column sum/sumsq stats. f16 input.
__global__ __launch_bounds__(256) void k_gemm_act_stats(
    const _Float16* in, const _Float16* __restrict__ wbt,
    const float* __restrict__ bias, _Float16* out,
    float* __restrict__ gstats, int ntiles) {
  __shared__ float statsS[256];
  int tid = threadIdx.x, wv = tid >> 5, lane = tid & 31;
  statsS[tid] = 0.0f;
  __syncthreads();
  int tile = blockIdx.x * 8 + wv;
  if (tile < ntiles) {
    int col0 = lane & 15;
    v8f acc[8];
#pragma unroll
    for (int nt = 0; nt < 8; ++nt) acc[nt] = bcast8(bias[nt * 16 + col0]);
    const _Float16* arow = in + (size_t)(tile * 16 + col0) * Hh;
    for (int kk = 0; kk < 4; ++kk) {
      v16h a = a_frag(arow, kk * 32, lane);
#pragma unroll
      for (int nt = 0; nt < 8; ++nt) WMMA_F16(acc[nt], a, b_frag(wbt, 128, nt * 16, kk * 32, lane));
    }
    int rb = tile * 16 + ((lane >> 4) << 3);
#pragma unroll
    for (int nt = 0; nt < 8; ++nt) {
      int col = nt * 16 + col0;
      float ps = 0.0f, pss = 0.0f;
#pragma unroll
      for (int i = 0; i < 8; ++i) {
        float z = acc[nt][i];
        z = z >= 0.0f ? z : 0.01f * z;
        ps += z; pss += z * z;
        out[(size_t)(rb + i) * Hh + col] = (_Float16)z;
      }
      atomicAdd(&statsS[col], ps);
      atomicAdd(&statsS[128 + col], pss);
    }
  }
  __syncthreads();
  atomicAdd(&gstats[tid], statsS[tid]);
}

// Same but f32 input (for z @ Wn1).
__global__ __launch_bounds__(256) void k_gemm_act_stats_f32(
    const float* __restrict__ in, const _Float16* __restrict__ wbt,
    const float* __restrict__ bias, _Float16* __restrict__ out,
    float* __restrict__ gstats, int ntiles) {
  __shared__ float statsS[256];
  int tid = threadIdx.x, wv = tid >> 5, lane = tid & 31;
  statsS[tid] = 0.0f;
  __syncthreads();
  int tile = blockIdx.x * 8 + wv;
  if (tile < ntiles) {
    int col0 = lane & 15;
    v8f acc[8];
#pragma unroll
    for (int nt = 0; nt < 8; ++nt) acc[nt] = bcast8(bias[nt * 16 + col0]);
    const float* arow = in + (size_t)(tile * 16 + col0) * Hh;
    for (int kk = 0; kk < 4; ++kk) {
      v16h a = a_frag_f32(arow, kk * 32, lane);
#pragma unroll
      for (int nt = 0; nt < 8; ++nt) WMMA_F16(acc[nt], a, b_frag(wbt, 128, nt * 16, kk * 32, lane));
    }
    int rb = tile * 16 + ((lane >> 4) << 3);
#pragma unroll
    for (int nt = 0; nt < 8; ++nt) {
      int col = nt * 16 + col0;
      float ps = 0.0f, pss = 0.0f;
#pragma unroll
      for (int i = 0; i < 8; ++i) {
        float z = acc[nt][i];
        z = z >= 0.0f ? z : 0.01f * z;
        ps += z; pss += z * z;
        out[(size_t)(rb + i) * Hh + col] = (_Float16)z;
      }
      atomicAdd(&statsS[col], ps);
      atomicAdd(&statsS[128 + col], pss);
    }
  }
  __syncthreads();
  atomicAdd(&gstats[tid], statsS[tid]);
}

// ---------------------------------------------------------------------------
// Node: xd = xh @ Wd + bd ; z = v * xd
__global__ __launch_bounds__(256) void k_node_xd(
    const _Float16* __restrict__ xh, const _Float16* __restrict__ Wdbt,
    const float* __restrict__ bd, const float* __restrict__ vparm,
    float* __restrict__ xd, float* __restrict__ z) {
  int tid = threadIdx.x, wv = tid >> 5, lane = tid & 31;
  int tile = blockIdx.x * 8 + wv;
  if (tile >= Nn / 16) return;
  int col0 = lane & 15;
  v8f acc[8];
#pragma unroll
  for (int nt = 0; nt < 8; ++nt) acc[nt] = bcast8(bd[nt * 16 + col0]);
  const _Float16* arow = xh + (size_t)(tile * 16 + col0) * Hh;
  for (int kk = 0; kk < 4; ++kk) {
    v16h a = a_frag(arow, kk * 32, lane);
#pragma unroll
    for (int nt = 0; nt < 8; ++nt) WMMA_F16(acc[nt], a, b_frag(Wdbt, 128, nt * 16, kk * 32, lane));
  }
  int rb = tile * 16 + ((lane >> 4) << 3);
#pragma unroll
  for (int nt = 0; nt < 8; ++nt) {
    int col = nt * 16 + col0;
    float vc = vparm[col];
#pragma unroll
    for (int i = 0; i < 8; ++i) {
      float t = acc[nt][i];
      xd[(size_t)(rb + i) * Hh + col] = t;
      z[(size_t)(rb + i) * Hh + col] = vc * t;
    }
  }
}

// ---------------------------------------------------------------------------
// Edge stage 3: cond = coeff * (a2 @ We3' + bias3'); m_st = cond * xd[src];
// z[dst] += m_st  (segment_sum via global float atomics).
__global__ __launch_bounds__(256) void k_edge3(
    const _Float16* __restrict__ a2, const _Float16* __restrict__ We3s,
    const float* __restrict__ bias3p, const float* __restrict__ ea,
    const int* __restrict__ ei, const float* __restrict__ xd,
    float* __restrict__ z) {
  __shared__ float coeffS[8][16];
  __shared__ int srcS[8][16], dstS[8][16];
  int tid = threadIdx.x, wv = tid >> 5, lane = tid & 31;
  int tile = blockIdx.x * 8 + wv;
  if (tile >= Ee / 16) return;
  if (lane < 16) {
    int e = tile * 16 + lane;
    coeffS[wv][lane] = __cosf(1.57079632679f * ea[(size_t)e * 4 + 3]);
    srcS[wv][lane] = ei[e];
    dstS[wv][lane] = ei[Ee + e];
  }
  asm volatile("s_wait_dscnt 0" ::: "memory");
  int col0 = lane & 15;
  v8f acc[8];
#pragma unroll
  for (int nt = 0; nt < 8; ++nt) acc[nt] = bcast8(bias3p[nt * 16 + col0]);
  const _Float16* arow = a2 + (size_t)(tile * 16 + col0) * Hh;
  for (int kk = 0; kk < 4; ++kk) {
    v16h a = a_frag(arow, kk * 32, lane);
#pragma unroll
    for (int nt = 0; nt < 8; ++nt) WMMA_F16(acc[nt], a, b_frag(We3s, 128, nt * 16, kk * 32, lane));
  }
  int j0 = (lane >> 4) << 3;
#pragma unroll
  for (int nt = 0; nt < 8; ++nt) {
    int col = nt * 16 + col0;
#pragma unroll
    for (int i = 0; i < 8; ++i) {
      int j = j0 + i;
      float val = coeffS[wv][j] * acc[nt][i] * xd[(size_t)srcS[wv][j] * Hh + col];
      atomicAdd(&z[(size_t)dstS[wv][j] * Hh + col], val);
    }
  }
}

// ---------------------------------------------------------------------------
// Node out: out = a3 @ Wn2' + biasN' + x
__global__ __launch_bounds__(256) void k_node_out(
    const _Float16* __restrict__ a3, const _Float16* __restrict__ wbt,
    const float* __restrict__ bias, const float* __restrict__ x,
    float* __restrict__ out) {
  int tid = threadIdx.x, wv = tid >> 5, lane = tid & 31;
  int tile = blockIdx.x * 8 + wv;
  if (tile >= Nn / 16) return;
  int col0 = lane & 15;
  v8f acc[8];
#pragma unroll
  for (int nt = 0; nt < 8; ++nt) acc[nt] = bcast8(bias[nt * 16 + col0]);
  const _Float16* arow = a3 + (size_t)(tile * 16 + col0) * Hh;
  for (int kk = 0; kk < 4; ++kk) {
    v16h a = a_frag(arow, kk * 32, lane);
#pragma unroll
    for (int nt = 0; nt < 8; ++nt) WMMA_F16(acc[nt], a, b_frag(wbt, 128, nt * 16, kk * 32, lane));
  }
  int rb = tile * 16 + ((lane >> 4) << 3);
#pragma unroll
  for (int nt = 0; nt < 8; ++nt) {
    int col = nt * 16 + col0;
#pragma unroll
    for (int i = 0; i < 8; ++i)
      out[(size_t)(rb + i) * Hh + col] = acc[nt][i] + x[(size_t)(rb + i) * Hh + col];
  }
}

// ---------------------------------------------------------------------------
// BN stats -> (scale s, shift t) per column.
__global__ void k_stats_fin(const float* __restrict__ gstats, float inv_cnt,
                            const float* __restrict__ g, const float* __restrict__ b,
                            float* __restrict__ s, float* __restrict__ t) {
  int n = threadIdx.x;
  float m = gstats[n] * inv_cnt;
  float var = gstats[128 + n] * inv_cnt - m * m;
  float sc = g[n] * rsqrtf(var + 1e-5f);
  s[n] = sc;
  t[n] = b[n] - m * sc;
}

// Fold bn(a)@W + bias into a@W' + bias': W'[n][k] = W[n][k]*s[k], bias' = bias + t@W.
__global__ __launch_bounds__(128) void k_fold(
    const _Float16* __restrict__ wbt, const float* __restrict__ s,
    const float* __restrict__ t, const float* __restrict__ bias_in,
    _Float16* __restrict__ wout, float* __restrict__ bias_out) {
  __shared__ float red[128];
  int n = blockIdx.x, k = threadIdx.x;
  float w = (float)wbt[(size_t)n * 128 + k];
  wout[(size_t)n * 128 + k] = (_Float16)(w * s[k]);
  red[k] = t[k] * w;
  __syncthreads();
  for (int o = 64; o > 0; o >>= 1) {
    if (k < o) red[k] += red[k + o];
    __syncthreads();
  }
  if (k == 0) bias_out[n] = bias_in[n] + red[0];
}

// ---------------------------------------------------------------------------
extern "C" void kernel_launch(void* const* d_in, const int* in_sizes, int n_in,
                              void* d_out, int out_size, void* d_ws, size_t ws_size,
                              hipStream_t stream) {
  (void)in_sizes; (void)n_in; (void)out_size; (void)ws_size;
  const float* x    = (const float*)d_in[0];
  const float* ea   = (const float*)d_in[1];
  const int*   ei   = (const int*)d_in[2];
  const float* Wb   = (const float*)d_in[3];
  const float* bb   = (const float*)d_in[4];
  const float* We1  = (const float*)d_in[5];
  const float* be1  = (const float*)d_in[6];
  const float* ge1  = (const float*)d_in[7];
  const float* bte1 = (const float*)d_in[8];
  const float* We2  = (const float*)d_in[9];
  const float* be2  = (const float*)d_in[10];
  const float* ge2  = (const float*)d_in[11];
  const float* bte2 = (const float*)d_in[12];
  const float* We3  = (const float*)d_in[13];
  const float* be3  = (const float*)d_in[14];
  const float* Wd   = (const float*)d_in[15];
  const float* bd   = (const float*)d_in[16];
  const float* vprm = (const float*)d_in[17];
  const float* Wn1  = (const float*)d_in[18];
  const float* bn1  = (const float*)d_in[19];
  const float* gn   = (const float*)d_in[20];
  const float* btn  = (const float*)d_in[21];
  const float* Wn2  = (const float*)d_in[22];
  const float* bn2b = (const float*)d_in[23];

  char* ws = (char*)d_ws;
  size_t off = 0;
  auto alloc = [&](size_t bytes) -> void* {
    void* p = ws + off;
    off = (off + bytes + 255) & ~(size_t)255;
    return p;
  };
  _Float16* XH    = (_Float16*)alloc((size_t)Nn * Hh * 2);  // x in f16 (reused as a3 later)
  _Float16* A1    = (_Float16*)alloc((size_t)Ee * Hh * 2);  // a1 then (in-place) a2
  float*    XD    = (float*)alloc((size_t)Nn * Hh * 4);
  float*    Z     = (float*)alloc((size_t)Nn * Hh * 4);
  _Float16* We1bt = (_Float16*)alloc((size_t)128 * 384 * 2);
  _Float16* Wbbt  = (_Float16*)alloc((size_t)128 * 224 * 2);
  _Float16* We2bt = (_Float16*)alloc((size_t)128 * 128 * 2);
  _Float16* We3bt = (_Float16*)alloc((size_t)128 * 128 * 2);
  _Float16* Wdbt  = (_Float16*)alloc((size_t)128 * 128 * 2);
  _Float16* Wn1bt = (_Float16*)alloc((size_t)128 * 128 * 2);
  _Float16* Wn2bt = (_Float16*)alloc((size_t)128 * 128 * 2);
  _Float16* We2s  = (_Float16*)alloc((size_t)128 * 128 * 2);
  _Float16* We3s  = (_Float16*)alloc((size_t)128 * 128 * 2);
  _Float16* Wn2s  = (_Float16*)alloc((size_t)128 * 128 * 2);
  float* stats1 = (float*)alloc(256 * 4);
  float* stats2 = (float*)alloc(256 * 4);
  float* stats3 = (float*)alloc(256 * 4);
  float* s1 = (float*)alloc(128 * 4); float* t1 = (float*)alloc(128 * 4);
  float* s2 = (float*)alloc(128 * 4); float* t2 = (float*)alloc(128 * 4);
  float* s3 = (float*)alloc(128 * 4); float* t3 = (float*)alloc(128 * 4);
  float* bias2p = (float*)alloc(128 * 4);
  float* bias3p = (float*)alloc(128 * 4);
  float* biasNp = (float*)alloc(128 * 4);
  _Float16* A3 = XH;  // xh no longer needed after edge1/node1

  hipMemsetAsync(stats1, 0, 3 * 256 * 4, stream);

  // Weight prep: f16, n-major (B-fragment friendly).
  k_prep_w<<<128, 128, 0, stream>>>(We1, We1bt, 384, 384);
  k_prep_w<<<128, 128, 0, stream>>>(Wb,  Wbbt,  200, 224);
  k_prep_w<<<128, 128, 0, stream>>>(We2, We2bt, 128, 128);
  k_prep_w<<<128, 128, 0, stream>>>(We3, We3bt, 128, 128);
  k_prep_w<<<128, 128, 0, stream>>>(Wd,  Wdbt,  128, 128);
  k_prep_w<<<128, 128, 0, stream>>>(Wn1, Wn1bt, 128, 128);
  k_prep_w<<<128, 128, 0, stream>>>(Wn2, Wn2bt, 128, 128);
  k_f32_to_f16<<<(Nn * Hh + 255) / 256, 256, 0, stream>>>(x, XH, Nn * Hh);

  // Edge stage 1 (fused RBF + Wb GEMM + We1 GEMM + lrelu + stats).
  k_edge1<<<(Ee / 16 + 3) / 4, 128, 0, stream>>>(XH, ea, ei, Wbbt, bb, We1bt, be1, A1, stats1);
  k_stats_fin<<<1, 128, 0, stream>>>(stats1, 1.0f / (float)Ee, ge1, bte1, s1, t1);
  k_fold<<<128, 128, 0, stream>>>(We2bt, s1, t1, be2, We2s, bias2p);

  // Edge stage 2 (BN folded into We2'); in-place a1 -> a2.
  k_gemm_act_stats<<<(Ee / 16 + 7) / 8, 256, 0, stream>>>(A1, We2s, bias2p, A1, stats2, Ee / 16);
  k_stats_fin<<<1, 128, 0, stream>>>(stats2, 1.0f / (float)Ee, ge2, bte2, s2, t2);
  k_fold<<<128, 128, 0, stream>>>(We3bt, s2, t2, be3, We3s, bias3p);

  // Node: xd = x@Wd+bd, z = v*xd (scatter target initialized here).
  k_node_xd<<<(Nn / 16 + 7) / 8, 256, 0, stream>>>(XH, Wdbt, bd, vprm, XD, Z);

  // Edge stage 3: cond filter + message + scatter-add into z.
  k_edge3<<<(Ee / 16 + 7) / 8, 256, 0, stream>>>(A1, We3s, bias3p, ea, ei, XD, Z);

  // Node MLP: a3 = lrelu(z@Wn1+bn1) + stats; fold BN; out = a3@Wn2' + bias' + x.
  k_gemm_act_stats_f32<<<(Nn / 16 + 7) / 8, 256, 0, stream>>>(Z, Wn1bt, bn1, A3, stats3, Nn / 16);
  k_stats_fin<<<1, 128, 0, stream>>>(stats3, 1.0f / (float)Nn, gn, btn, s3, t3);
  k_fold<<<128, 128, 0, stream>>>(Wn2bt, s3, t3, bn2b, Wn2s, biasNp);
  k_node_out<<<(Nn / 16 + 7) / 8, 256, 0, stream>>>(A3, Wn2s, biasNp, x, (float*)d_out);
}
